// WinAttn_57114475102241
// MI455X (gfx1250) — matrix operand order repaired and checked by reading, not compiled, not run on hardware
//
#include <hip/hip_runtime.h>

typedef __bf16 bf16;
typedef __attribute__((ext_vector_type(8)))  __bf16 v8bf;
typedef __attribute__((ext_vector_type(16))) __bf16 v16bf;
typedef __attribute__((ext_vector_type(8)))  float  v8f;

// ---------------------------------------------------------------------------
// WMMA helpers (CDNA5 wave32, V_WMMA_F32_16X16X32_BF16)
// A fragment (16x32, 16-bit): lane l holds row M=l%16; element j = K j+16*(l>=16) (this code's k order).
// B fragment (32x16, 16-bit): lane l holds col N=l%16; element j = K j+16*(l>=16) (this code's k order, same as A).
// D fragment (16x16 f32):     lane l holds col N=l%16; vgpr r = row M r+8*(l>=16).
// ---------------------------------------------------------------------------
__device__ __forceinline__ v16bf load_a_frag(const bf16* p) {
  // p = &src[row * stride + k0 + 16*half]   (16-byte aligned)
  const v8bf* q = (const v8bf*)p;
  v8bf lo = q[0];      // K = base .. base+7
  v8bf hi = q[1];      // K = base+8 .. base+15
  return __builtin_shufflevector(lo, hi, 0,1,2,3,4,5,6,7,8,9,10,11,12,13,14,15);
}
__device__ __forceinline__ v16bf load_b_frag(const bf16* p) {
  // p = &src[col_row * stride + k0 + 16*half]   (16 contiguous bf16)
  const v8bf* q = (const v8bf*)p;
  return __builtin_shufflevector(q[0], q[1], 0,1,2,3,4,5,6,7,8,9,10,11,12,13,14,15);
}
__device__ __forceinline__ v8f wmma_bf16(v16bf a, v16bf b, v8f c) {
  return __builtin_amdgcn_wmma_f32_16x16x32_bf16(false, a, false, b, (short)0, c,
                                                 false, false);
}

// ---------------------------------------------------------------------------
// Prep: convert + transpose weights to bf16 in workspace.
//   wqkvT[oc][ic] = w_qkv[ic][oc]   (1536 x 512)
//   wpT  [oc][ic] = w_proj[ic][oc]  ( 512 x 512)
// ---------------------------------------------------------------------------
__global__ __launch_bounds__(256) void winattn_prep(
    const float* __restrict__ w_qkv, const float* __restrict__ w_proj,
    bf16* __restrict__ wqkvT, bf16* __restrict__ wpT) {
  int idx = blockIdx.x * 256 + threadIdx.x;
  if (idx < 1536 * 512) {
    int oc = idx >> 9, ic = idx & 511;
    wqkvT[idx] = (bf16)w_qkv[ic * 1536 + oc];
  } else {
    int j = idx - 1536 * 512;
    if (j < 512 * 512) {
      int oc = j >> 9, ic = j & 511;
      wpT[j] = (bf16)w_proj[ic * 512 + oc];
    }
  }
}

// ---------------------------------------------------------------------------
// Fused window attention: one workgroup (256 thr = 8 waves) per window.
// ---------------------------------------------------------------------------
#define XS 520   // x_lds row stride (bf16 elems)
#define OS 520   // o_lds row stride
#define QS 40    // q/k row stride
#define VS 72    // vT row stride
#define SS 68    // s row stride (f32)
#define PS 136   // p row stride (bf16, overlays s)

#define OFF_X  0
#define OFF_O  66560
#define OFF_R  133120          // q | k  -> later s | p (region reused)
#define OFF_K  (OFF_R + 5120)
#define OFF_VT 150528
#define SMEM_BYTES 155136

__global__ __launch_bounds__(256) void winattn_main(
    const float* __restrict__ x, const bf16* __restrict__ wqkvT,
    const float* __restrict__ bqkv, const bf16* __restrict__ wpT,
    const float* __restrict__ bproj, float* __restrict__ out) {
  __shared__ __align__(128) char smem[SMEM_BYTES];
  bf16*  x_lds  = (bf16*)(smem + OFF_X);
  bf16*  o_lds  = (bf16*)(smem + OFF_O);
  bf16*  q_lds  = (bf16*)(smem + OFF_R);
  bf16*  k_lds  = (bf16*)(smem + OFF_K);
  float* s_lds  = (float*)(smem + OFF_R);
  bf16*  p_lds  = (bf16*)(smem + OFF_R);
  bf16*  vT_lds = (bf16*)(smem + OFF_VT);

  const int tid = threadIdx.x;
  const int lid = tid & 31;
  const int wid = tid >> 5;
  const int hl  = lid >> 4;     // half-wave: 0 or 1
  const int l16 = lid & 15;

  const int widx = blockIdx.x;
  const int bb = widx >> 6;           // batch
  const int wy = (widx >> 3) & 7;     // window row
  const int wx = widx & 7;            // window col

  // ---- Stage A: load window of x (64 tokens x 512 ch), f32 -> bf16, zero-pad
  for (int it = 0; it < 32; ++it) {
    int idx = it * 256 + tid;         // 8192 float4 groups
    int t   = idx >> 7;               // token 0..63
    int ch  = (idx & 127) << 2;       // channel 0..508 step 4
    int gh  = wy * 8 + (t >> 3);
    int gw  = wx * 8 + (t & 7);
    float4 val = make_float4(0.f, 0.f, 0.f, 0.f);
    if (gh < 60 && gw < 60)
      val = *(const float4*)(x + ((size_t)(bb * 3600 + gh * 60 + gw) * 512 + ch));
    bf16* dst = x_lds + t * XS + ch;
    dst[0] = (bf16)val.x; dst[1] = (bf16)val.y;
    dst[2] = (bf16)val.z; dst[3] = (bf16)val.w;
  }
  __syncthreads();

  const float scale = 0.17677669529663687f;   // 32^-0.5
  const int rb = wid >> 1, cb = wid & 1;      // 4x2 tile grid for q/k/v/o
  const int m0 = rb * 16, n0 = cb * 16;

  for (int h = 0; h < 16; ++h) {
    // ---- Q/K/V = x @ w_qkv(+bias): one A-fragment feeds 3 WMMAs (q,k,v)
    {
      int qcol = h * 32 + n0 + l16;             // q column in w_qkv
      v8f accq, acck, accv;
      float bq = bqkv[qcol];
      float bk = bqkv[512 + qcol];
      float bv = bqkv[1024 + qcol];
#pragma unroll
      for (int r = 0; r < 8; ++r) { accq[r] = bq; acck[r] = bk; accv[r] = bv; }
      const bf16* wq = wqkvT + (size_t)qcol * 512;
      const bf16* wk = wq + (size_t)512 * 512;
      const bf16* wv = wq + (size_t)1024 * 512;
#pragma unroll
      for (int kk = 0; kk < 16; ++kk) {
        v16bf a = load_a_frag(x_lds + (m0 + l16) * XS + kk * 32 + 16 * hl);
        accq = wmma_bf16(a, load_b_frag(wq + kk * 32 + 16 * hl), accq);
        acck = wmma_bf16(a, load_b_frag(wk + kk * 32 + 16 * hl), acck);
        accv = wmma_bf16(a, load_b_frag(wv + kk * 32 + 16 * hl), accv);
      }
#pragma unroll
      for (int r = 0; r < 8; ++r) {
        q_lds[(m0 + r + 8 * hl) * QS + n0 + l16] = (bf16)accq[r];
        k_lds[(m0 + r + 8 * hl) * QS + n0 + l16] = (bf16)acck[r];
        vT_lds[(n0 + l16) * VS + m0 + r + 8 * hl] = (bf16)accv[r];   // V^T
      }
    }
    __syncthreads();

    // ---- S = (q @ k^T) * scale : 16 tiles, 2 per wave sharing one A-frag
    int sm0 = (wid >> 1) * 16;          // row-block of this wave's S tiles
    int snb = (wid & 1) * 32;           // base of its two column tiles
    v8f sacc[2];
    {
      v16bf a = load_a_frag(q_lds + (sm0 + l16) * QS + 16 * hl);
#pragma unroll
      for (int ti = 0; ti < 2; ++ti) {
        v16bf bf = load_b_frag(k_lds + (snb + ti * 16 + l16) * QS + 16 * hl);
        v8f z = {};
        sacc[ti] = wmma_bf16(a, bf, z);
      }
    }
    __syncthreads();   // all waves done reading q/k; region reused for s
#pragma unroll
    for (int ti = 0; ti < 2; ++ti) {
#pragma unroll
      for (int r = 0; r < 8; ++r)
        s_lds[(sm0 + r + 8 * hl) * SS + snb + ti * 16 + l16] = sacc[ti][r] * scale;
    }
    __syncthreads();

    // ---- row softmax (64 rows, one thread each), write P (bf16) in place
    if (tid < 64) {
      const float* srow = s_lds + tid * SS;
      float vals[64];
      float mx = -3.0e38f;
#pragma unroll
      for (int j = 0; j < 64; ++j) { vals[j] = srow[j]; mx = fmaxf(mx, vals[j]); }
      float sum = 0.f;
#pragma unroll
      for (int j = 0; j < 64; ++j) { vals[j] = __expf(vals[j] - mx); sum += vals[j]; }
      float inv = 1.0f / sum;
      bf16* prow = p_lds + tid * PS;
#pragma unroll
      for (int j = 0; j < 64; ++j) prow[j] = (bf16)(vals[j] * inv);
    }
    __syncthreads();

    // ---- O_h = P @ V : one 16x16 tile per wave, K=64 -> 2 WMMAs
    {
      v8f acc = {};
#pragma unroll
      for (int c = 0; c < 2; ++c) {
        v16bf a  = load_a_frag(p_lds + (m0 + l16) * PS + c * 32 + 16 * hl);
        v16bf bf = load_b_frag(vT_lds + (n0 + l16) * VS + c * 32 + 16 * hl);
        acc = wmma_bf16(a, bf, acc);
      }
#pragma unroll
      for (int r = 0; r < 8; ++r)
        o_lds[(m0 + r + 8 * hl) * OS + h * 32 + n0 + l16] = (bf16)acc[r];
    }
    __syncthreads();
  }

  // ---- Projection: out = o @ w_proj + b_proj, fused window-reverse + crop.
  // Wave owns 4 consecutive column tiles; one A-fragment feeds 4 WMMAs.
  {
    const int cb0 = wid * 4;                 // column tiles cb0..cb0+3
    float bias[4];
    const bf16* wrow[4];
#pragma unroll
    for (int c = 0; c < 4; ++c) {
      int col = (cb0 + c) * 16 + l16;
      bias[c] = bproj[col];
      wrow[c] = wpT + (size_t)col * 512;
    }
    for (int prb = 0; prb < 4; ++prb) {
      int pm0 = prb * 16;
      v8f acc[4];
#pragma unroll
      for (int c = 0; c < 4; ++c)
#pragma unroll
        for (int r = 0; r < 8; ++r) acc[c][r] = bias[c];
#pragma unroll
      for (int kk = 0; kk < 16; ++kk) {
        v16bf a = load_a_frag(o_lds + (pm0 + l16) * OS + kk * 32 + 16 * hl);
#pragma unroll
        for (int c = 0; c < 4; ++c)
          acc[c] = wmma_bf16(a, load_b_frag(wrow[c] + kk * 32 + 16 * hl), acc[c]);
      }
#pragma unroll
      for (int r = 0; r < 8; ++r) {
        int tok = pm0 + r + 8 * hl;
        int gh  = wy * 8 + (tok >> 3);
        int gw  = wx * 8 + (tok & 7);
        if (gh < 60 && gw < 60) {
          float* orow = out + (size_t)(bb * 3600 + gh * 60 + gw) * 512;
#pragma unroll
          for (int c = 0; c < 4; ++c)
            orow[(cb0 + c) * 16 + l16] = acc[c][r];
        }
      }
    }
  }
}

// ---------------------------------------------------------------------------
extern "C" void kernel_launch(void* const* d_in, const int* in_sizes, int n_in,
                              void* d_out, int out_size, void* d_ws, size_t ws_size,
                              hipStream_t stream) {
  const float* x      = (const float*)d_in[0];
  const float* w_qkv  = (const float*)d_in[1];
  const float* b_qkv  = (const float*)d_in[2];
  const float* w_proj = (const float*)d_in[3];
  const float* b_proj = (const float*)d_in[4];
  // d_in[5]=H, d_in[6]=W (both 60, fixed at compile time)

  bf16* wqkvT = (bf16*)d_ws;                                      // 1536*512 bf16
  bf16* wpT   = (bf16*)((char*)d_ws + 1536 * 512 * sizeof(bf16)); // 512*512 bf16

  winattn_prep<<<4096, 256, 0, stream>>>(w_qkv, w_proj, wqkvT, wpT);
  winattn_main<<<2048, 256, 0, stream>>>(x, wqkvT, b_qkv, wpT, b_proj,
                                         (float*)d_out);
}
